// RetNetDecoder_57131654971866
// MI455X (gfx1250) — matrix operand order repaired and checked by reading, not compiled, not run on hardware
//
#include <hip/hip_runtime.h>
#include <cmath>

typedef __bf16 bf16;
typedef __attribute__((ext_vector_type(16))) __bf16 v16bf;
typedef __attribute__((ext_vector_type(8)))  __bf16 bf16x8;
typedef __attribute__((ext_vector_type(8)))  float  v8f;
typedef __attribute__((ext_vector_type(4)))  unsigned v4u;
typedef __attribute__((ext_vector_type(8)))  int      v8i;
typedef __attribute__((ext_vector_type(4)))  int      v4i;

namespace cfg {
constexpr int B = 2, T = 2048, C = 512, H = 8, L = 4, FFN = 2048;
constexpr int KD = C / H;        // 64
constexpr int BT = B * T;        // 4096 rows
constexpr float SCALE = 0.125f;  // KD^-0.5
constexpr float EPS = 1e-6f;
}

__device__ __forceinline__ float gelu_exact(float x) {
  return 0.5f * x * (1.0f + erff(x * 0.70710678118654752f));
}

__device__ __forceinline__ v8f vzero8() {
  v8f z;
#pragma unroll
  for (int i = 0; i < 8; ++i) z[i] = 0.0f;
  return z;
}

__device__ __forceinline__ v8f wmma_bf16(v16bf a, v16bf b, v8f c) {
  return __builtin_amdgcn_wmma_f32_16x16x32_bf16(false, a, false, b, (short)0, c,
                                                 false, false);
}

// LDS byte offset of a __shared__ object (generic -> AS(3) -> int)
__device__ __forceinline__ unsigned lds_off(void* p) {
  return (unsigned)(uintptr_t)(__attribute__((address_space(3))) char*)p;
}

// ---------------------------------------------------------------- TDM 2D tile load
// Loads a (tile_h x tile_w) tile of bf16 (2-byte) elements with row stride
// `row_stride` (elements) from `gaddr` into LDS at byte offset `ldsoff`,
// packed contiguously (row-major, tile_w elements per row).
__device__ __forceinline__ void tdm_load_2d(unsigned ldsoff, const void* gaddr,
                                            unsigned tile_w, unsigned tile_h,
                                            unsigned row_stride) {
  const unsigned long long ga = (unsigned long long)(uintptr_t)gaddr;
  v4u g0;
  g0[0] = 1u;                                        // count=1 (valid), user mode
  g0[1] = ldsoff;                                    // lds_addr (bytes)
  g0[2] = (unsigned)(ga & 0xFFFFFFFFu);              // global_addr[31:0]
  g0[3] = (unsigned)((ga >> 32) & 0x1FFFFFFu) | (2u << 30);  // addr[56:32] | type=2
  const unsigned dim0 = row_stride;                  // tensor_dim0 (elements)
  const unsigned dim1 = tile_h;                      // tensor_dim1
  v8i g1;
  g1[0] = (int)(1u << 16);                           // data_size=1 -> 2 bytes
  g1[1] = (int)((dim0 & 0xFFFFu) << 16);             // tensor_dim0[15:0] @63:48
  g1[2] = (int)((dim0 >> 16) | ((dim1 & 0xFFFFu) << 16));  // dim0[31:16] | dim1[15:0]
  g1[3] = (int)((dim1 >> 16) | (tile_w << 16));      // dim1[31:16] | tile_dim0 @127:112
  g1[4] = (int)(tile_h & 0xFFFFu);                   // tile_dim1 @143:128 (tile_dim2=0)
  g1[5] = (int)row_stride;                           // tensor_dim0_stride[31:0] @191:160
  g1[6] = 0;                                         // stride0[47:32] | dim1_stride lo
  g1[7] = 0;
  const v4i z4 = {0, 0, 0, 0};
  const v8i z8 = {0, 0, 0, 0, 0, 0, 0, 0};
  // 6-arg toolchain form: (g0, g1, g2, g3, extra, cpol)
  __builtin_amdgcn_tensor_load_to_lds(g0, g1, z4, z4, z8, 0);
}

// ---------------------------------------------------------------- copy
__global__ void copy_f32(const float* __restrict__ src, float* __restrict__ dst, int n) {
  int i = blockIdx.x * blockDim.x + threadIdx.x;
  if (i < n) dst[i] = src[i];
}

// ---------------------------------------------------------------- W[K,N] f32 -> Wt[N,K] bf16
__launch_bounds__(256)
__global__ void transpose_cvt_bf16(const float* __restrict__ W, bf16* __restrict__ Wt,
                                   int K, int N) {
  __shared__ bf16 tile[32][33];
  const int k0 = blockIdx.x * 32, n0 = blockIdx.y * 32;
  const int tx = threadIdx.x & 31, ty = threadIdx.x >> 5;  // 8 rows per pass
#pragma unroll
  for (int r = 0; r < 32; r += 8)
    tile[ty + r][tx] = (bf16)W[(size_t)(k0 + ty + r) * N + n0 + tx];
  __syncthreads();
#pragma unroll
  for (int r = 0; r < 32; r += 8)
    Wt[(size_t)(n0 + ty + r) * K + k0 + tx] = tile[tx][ty + r];
}

// ---------------------------------------------------------------- LayerNorm -> bf16
__launch_bounds__(256)
__global__ void ln_to_bf16(const float* __restrict__ in, bf16* __restrict__ out, int W) {
  __shared__ float red[256];
  const int tid = threadIdx.x;
  const float* p = in + (size_t)blockIdx.x * W;
  float s = 0.f;
  for (int i = tid; i < W; i += 256) s += p[i];
  red[tid] = s; __syncthreads();
  for (int o = 128; o > 0; o >>= 1) { if (tid < o) red[tid] += red[tid + o]; __syncthreads(); }
  const float mean = red[0] * (1.0f / W);
  __syncthreads();
  float v = 0.f;
  for (int i = tid; i < W; i += 256) { float d = p[i] - mean; v += d * d; }
  red[tid] = v; __syncthreads();
  for (int o = 128; o > 0; o >>= 1) { if (tid < o) red[tid] += red[tid + o]; __syncthreads(); }
  const float inv = rsqrtf(red[0] * (1.0f / W) + cfg::EPS);
  bf16* q = out + (size_t)blockIdx.x * W;
  for (int i = tid; i < W; i += 256) q[i] = (bf16)((p[i] - mean) * inv);
}

// ---------------------------------------------------------------- GEMM (TDM double-buffered, bf16 WMMA)
// out[M,N] = (res? res : 0) + gelu(A_bf16[M,K] @ Wt_bf16[N,K]^T + bias)
__launch_bounds__(256)
__global__ void gemm_bf16_wmma(const bf16* __restrict__ A, const bf16* __restrict__ Wt,
                               const float* __restrict__ bias, float* __restrict__ out,
                               const float* __restrict__ res, int M, int N, int K) {
  __shared__ __align__(16) bf16 At[2][128][32];  // 16 KB
  __shared__ __align__(16) bf16 Bt[2][64][32];   // 8 KB, [n][k]
  const int tid = threadIdx.x;
  const int lane = tid & 31;
  const int wave = tid >> 5;
  const int m0 = blockIdx.x * 128;
  const int n0 = blockIdx.y * 64;

  v8f acc[4];
#pragma unroll
  for (int c = 0; c < 4; ++c) acc[c] = vzero8();

  const int nk = K / 32;
  if (wave == 0) {  // prologue: DMA first tiles
    tdm_load_2d(lds_off(&At[0][0][0]), A + (size_t)m0 * K, 32, 128, (unsigned)K);
    tdm_load_2d(lds_off(&Bt[0][0][0]), Wt + (size_t)n0 * K, 32, 64, (unsigned)K);
  }

  for (int i = 0; i < nk; ++i) {
    if (wave == 0) __builtin_amdgcn_s_wait_tensorcnt(0);  // tile i resident
    __syncthreads();
    if (wave == 0 && i + 1 < nk) {  // DMA tile i+1 into the other buffer
      const int kn = (i + 1) * 32;
      tdm_load_2d(lds_off(&At[(i + 1) & 1][0][0]), A + (size_t)m0 * K + kn, 32, 128,
                  (unsigned)K);
      tdm_load_2d(lds_off(&Bt[(i + 1) & 1][0][0]), Wt + (size_t)n0 * K + kn, 32, 64,
                  (unsigned)K);
    }
    const bf16 (*Ab)[32] = At[i & 1];
    const bf16 (*Bb)[32] = Bt[i & 1];
    v16bf af;
    {  // A fragment: m = lane&15, k = (e>>3)*16 + (lane>>4)*8 + (e&7)
      const int m = wave * 16 + (lane & 15);
      const int kb = (lane >> 4) * 8;
      bf16x8 lo = *(const bf16x8*)&Ab[m][kb];
      bf16x8 hi = *(const bf16x8*)&Ab[m][16 + kb];
#pragma unroll
      for (int e = 0; e < 8; ++e) { af[e] = lo[e]; af[8 + e] = hi[e]; }
    }
#pragma unroll
    for (int c = 0; c < 4; ++c) {  // B fragment: n = lane&15, k = (e>>3)*16 + (lane>>4)*8 + (e&7)
      const int n = c * 16 + (lane & 15);
      const int kb = (lane >> 4) * 8;
      v16bf bfr;
      bf16x8 lo = *(const bf16x8*)&Bb[n][kb];
      bf16x8 hi = *(const bf16x8*)&Bb[n][16 + kb];
#pragma unroll
      for (int e = 0; e < 8; ++e) { bfr[e] = lo[e]; bfr[8 + e] = hi[e]; }
      acc[c] = wmma_bf16(af, bfr, acc[c]);
    }
    __syncthreads();  // everyone done with buffer i&1 before it is re-filled
  }

  // epilogue: D layout m = (lane>>4)*8 + r, n = lane&15
  const int mbase = m0 + wave * 16 + (lane >> 4) * 8;
  const int ncol  = n0 + (lane & 15);
#pragma unroll
  for (int c = 0; c < 4; ++c) {
    const int col = ncol + c * 16;
    const float bv = bias[col];
#pragma unroll
    for (int r = 0; r < 8; ++r) {
      const size_t idx = (size_t)(mbase + r) * N + col;
      float v = gelu_exact(acc[c][r] + bv);
      if (res) v += res[idx];
      out[idx] = v;
    }
  }
}

// ---------------------------------------------------------------- rotary / theta shift prep -> bf16
__global__ void rope_prep(const float* __restrict__ q, const float* __restrict__ k,
                          const float* __restrict__ v, bf16* __restrict__ qr,
                          bf16* __restrict__ kr, bf16* __restrict__ vr) {
  const int idx = blockIdx.x * blockDim.x + threadIdx.x;  // 2^20 pairs
  const int d2 = idx & 31;
  const int h  = (idx >> 5) & 7;
  const int t  = (idx >> 8) & (cfg::T - 1);
  const int b  = idx >> 19;
  const float ang = __powf(10000.0f, -(float)d2 * (1.0f / 31.0f));
  const float th = (float)t * ang;
  const float cs = cosf(th), sn = sinf(th);
  const size_t base = ((size_t)(b * cfg::T + t)) * cfg::C + h * cfg::KD + 2 * d2;
  const float q0 = q[base], q1 = q[base + 1];
  qr[base]     = (bf16)(q0 * cs - q1 * sn);
  qr[base + 1] = (bf16)(q1 * cs + q0 * sn);
  const float k0 = k[base] * cfg::SCALE, k1 = k[base + 1] * cfg::SCALE;
  kr[base]     = (bf16)(k0 * cs - k1 * sn);
  kr[base + 1] = (bf16)(k1 * cs + k0 * sn);
  vr[base]     = (bf16)v[base];
  vr[base + 1] = (bf16)v[base + 1];
}

// ---------------------------------------------------------------- retention (flash-style, WMMA + TDM v tiles)
__launch_bounds__(32)
__global__ void retention_wmma(const bf16* __restrict__ qr, const bf16* __restrict__ kr,
                               const bf16* __restrict__ vr, float* __restrict__ out) {
  __shared__ __align__(16) bf16 qkT[16][32];     // masked qk tile (A re-shape)
  __shared__ __align__(16) bf16 vt[2][32][64];   // double-buffered v s-chunks
  const int lane = threadIdx.x;
  const int t0 = blockIdx.x * 16;
  const int bh = blockIdx.y;
  const int b = bh >> 3, h = bh & 7;
  const float decay = __logf(1.0f - exp2f(-5.0f - (float)h));
  const float rr = __expf(decay);

  // loop-invariant qr A fragments (16x64 -> 2 k-chunks of 32)
  v16bf qa[2];
  {
    const int m = lane & 15;
    const int kb = (lane >> 4) * 8;
    const bf16* rowp = qr + ((size_t)(b * cfg::T + t0 + m)) * cfg::C + h * cfg::KD;
#pragma unroll
    for (int kc = 0; kc < 2; ++kc) {
      bf16x8 lo = *(const bf16x8*)(rowp + kc * 32 + kb);
      bf16x8 hi = *(const bf16x8*)(rowp + kc * 32 + 16 + kb);
#pragma unroll
      for (int e = 0; e < 8; ++e) { qa[kc][e] = lo[e]; qa[kc][8 + e] = hi[e]; }
    }
  }
  // mask row normalizer: sum_{s<=t} exp((t-s)*decay) = (1-r^{t+1})/(1-r)
  float invgeo[8];
#pragma unroll
  for (int r = 0; r < 8; ++r) {
    const int t = t0 + (lane >> 4) * 8 + r;
    const float geo = (1.0f - __expf((float)(t + 1) * decay)) / (1.0f - rr);
    invgeo[r] = rsqrtf(geo);
  }

  v8f oacc[4];
#pragma unroll
  for (int c = 0; c < 4; ++c) oacc[c] = vzero8();
  float rowsum[8];
#pragma unroll
  for (int r = 0; r < 8; ++r) rowsum[r] = 0.0f;

  const bf16* vbase = vr + ((size_t)(b * cfg::T)) * cfg::C + h * cfg::KD;
  const int nS = (t0 + 15) / 32 + 1;  // number of 32-wide s tiles
  tdm_load_2d(lds_off(&vt[0][0][0]), vbase, 64, 32, (unsigned)cfg::C);

  for (int i = 0; i < nS; ++i) {
    const int s0 = i * 32;
    if (i + 1 < nS) {  // overlap DMA of next v tile with this step's compute
      tdm_load_2d(lds_off(&vt[(i + 1) & 1][0][0]), vbase + (size_t)(s0 + 32) * cfg::C,
                  64, 32, (unsigned)cfg::C);
      __builtin_amdgcn_s_wait_tensorcnt(1);
    } else {
      __builtin_amdgcn_s_wait_tensorcnt(0);
    }

#pragma unroll
    for (int sub = 0; sub < 2; ++sub) {  // two 16-wide s subtiles
      const int sbase = s0 + sub * 16;
      v8f qk = vzero8();
      const bf16* krow = kr + ((size_t)(b * cfg::T + sbase + (lane & 15))) * cfg::C
                         + h * cfg::KD + (lane >> 4) * 8;
#pragma unroll
      for (int kc = 0; kc < 2; ++kc) {
        v16bf bfr;
        bf16x8 lo = *(const bf16x8*)(krow + kc * 32);
        bf16x8 hi = *(const bf16x8*)(krow + kc * 32 + 16);
#pragma unroll
        for (int e = 0; e < 8; ++e) { bfr[e] = lo[e]; bfr[8 + e] = hi[e]; }
        qk = wmma_bf16(qa[kc], bfr, qk);
      }
      // decay mask + rowsum accumulation + LDS round-trip (D -> A re-shape)
      const int mB = (lane >> 4) * 8;
      const int n  = lane & 15;
      const int s  = sbase + n;
#pragma unroll
      for (int r = 0; r < 8; ++r) {
        const int t = t0 + mB + r;
        const int d = t - s;
        const float w = (d >= 0) ? __expf((float)d * decay) * invgeo[r] : 0.0f;
        const float val = qk[r] * w;
        rowsum[r] += val;
        qkT[mB + r][sub * 16 + n] = (bf16)val;
      }
    }
    __syncthreads();

    // out += qk @ v (K = 32 s-values)
    v16bf af;
    {
      const int m = lane & 15;
      const int kb = (lane >> 4) * 8;
      bf16x8 lo = *(const bf16x8*)&qkT[m][kb];
      bf16x8 hi = *(const bf16x8*)&qkT[m][16 + kb];
#pragma unroll
      for (int e = 0; e < 8; ++e) { af[e] = lo[e]; af[8 + e] = hi[e]; }
    }
    const bf16 (*vb)[64] = vt[i & 1];
#pragma unroll
    for (int c = 0; c < 4; ++c) {
      const int n = c * 16 + (lane & 15);
      v16bf bfr;
#pragma unroll
      for (int e = 0; e < 16; ++e) bfr[e] = vb[(lane >> 4) * 8 + (e & 7) + (e >> 3) * 16][n];
      oacc[c] = wmma_bf16(af, bfr, oacc[c]);
    }
    __syncthreads();
  }

  // rowsum reduce across the 16-lane half holding each row (wave32)
#pragma unroll
  for (int r = 0; r < 8; ++r) {
    float v = rowsum[r];
    v += __shfl_xor(v, 1, 32);
    v += __shfl_xor(v, 2, 32);
    v += __shfl_xor(v, 4, 32);
    v += __shfl_xor(v, 8, 32);
    rowsum[r] = v;
  }

  const size_t obase = ((size_t)(b * cfg::T + t0)) * cfg::C + h * cfg::KD;
#pragma unroll
  for (int r = 0; r < 8; ++r) {
    const float inv = 1.0f / (fmaxf(fabsf(rowsum[r]), 1.0f) + cfg::EPS);
    const int m = (lane >> 4) * 8 + r;
#pragma unroll
    for (int c = 0; c < 4; ++c)
      out[obase + (size_t)m * cfg::C + c * 16 + (lane & 15)] = oacc[c][r] * inv;
  }
}

// ---------------------------------------------------------------- LN + silu(g) gating -> bf16
__launch_bounds__(256)
__global__ void gate_ln(const float* __restrict__ ret, const float* __restrict__ g,
                        bf16* __restrict__ out) {
  __shared__ float red[256];
  const int tid = threadIdx.x;
  const size_t row = blockIdx.x;
  const float* p = ret + row * cfg::C;
  const float* gp = g + row * cfg::C;
  float s = 0.f;
  for (int i = tid; i < cfg::C; i += 256) s += p[i];
  red[tid] = s; __syncthreads();
  for (int o = 128; o > 0; o >>= 1) { if (tid < o) red[tid] += red[tid + o]; __syncthreads(); }
  const float mean = red[0] * (1.0f / cfg::C);
  __syncthreads();
  float v = 0.f;
  for (int i = tid; i < cfg::C; i += 256) { float d = p[i] - mean; v += d * d; }
  red[tid] = v; __syncthreads();
  for (int o = 128; o > 0; o >>= 1) { if (tid < o) red[tid] += red[tid + o]; __syncthreads(); }
  const float inv = rsqrtf(red[0] * (1.0f / cfg::C) + cfg::EPS);
  bf16* q = out + row * cfg::C;
  for (int i = tid; i < cfg::C; i += 256) {
    const float gv = gp[i];
    const float silu = gv / (1.0f + __expf(-gv));
    q[i] = (bf16)(silu * ((p[i] - mean) * inv));
  }
}

// ---------------------------------------------------------------- host
extern "C" void kernel_launch(void* const* d_in, const int* in_sizes, int n_in,
                              void* d_out, int out_size, void* d_ws, size_t ws_size,
                              hipStream_t stream) {
  using namespace cfg;
  const float* x_in = (const float*)d_in[0];
  const float* Wq = (const float*)d_in[1];  const float* bq = (const float*)d_in[2];
  const float* Wk = (const float*)d_in[3];  const float* bk = (const float*)d_in[4];
  const float* Wv = (const float*)d_in[5];  const float* bv = (const float*)d_in[6];
  const float* Wg = (const float*)d_in[7];  const float* bg = (const float*)d_in[8];
  const float* Wo = (const float*)d_in[9];  const float* bo = (const float*)d_in[10];
  const float* W1 = (const float*)d_in[11]; const float* b1 = (const float*)d_in[12];
  const float* W2 = (const float*)d_in[13]; const float* b2 = (const float*)d_in[14];

  // workspace carve (~90 MB)
  char* ws = (char*)d_ws;
  float* f32buf = (float*)(ws);                       // 32 MB: q/k/v/g quarters, later FFN f32
  float* q32 = f32buf;
  float* k32 = f32buf + (size_t)1 * BT * C;
  float* v32 = f32buf + (size_t)2 * BT * C;
  float* g32 = f32buf + (size_t)3 * BT * C;
  float* ret32 = q32;                                 // alias: q32 dead after rope_prep
  bf16* hb  = (bf16*)(ws + ((size_t)32 << 20));       // 16 MB (holds up to BT x FFN bf16)
  bf16* qrb = (bf16*)(ws + ((size_t)48 << 20));
  bf16* krb = (bf16*)(ws + ((size_t)52 << 20));
  bf16* vrb = (bf16*)(ws + ((size_t)56 << 20));
  bf16* gtb = (bf16*)(ws + ((size_t)60 << 20));
  bf16* wpool = (bf16*)(ws + ((size_t)64 << 20));     // 26 MB bf16 transposed weights
  const size_t WCC = (size_t)C * C, WCF = (size_t)C * FFN;
  const size_t LSTRIDE = 5 * WCC + 2 * WCF;           // per-layer bf16 elements

  float* x = (float*)d_out;  // residual stream lives in d_out
  const int M = BT;
  const dim3 blk256(256);

  // one-time: convert + transpose all weights to bf16 [N,K]
  for (int l = 0; l < L; ++l) {
    bf16* base = wpool + (size_t)l * LSTRIDE;
    transpose_cvt_bf16<<<dim3(C / 32, C / 32), blk256, 0, stream>>>(Wq + (size_t)l * WCC, base + 0 * WCC, C, C);
    transpose_cvt_bf16<<<dim3(C / 32, C / 32), blk256, 0, stream>>>(Wk + (size_t)l * WCC, base + 1 * WCC, C, C);
    transpose_cvt_bf16<<<dim3(C / 32, C / 32), blk256, 0, stream>>>(Wv + (size_t)l * WCC, base + 2 * WCC, C, C);
    transpose_cvt_bf16<<<dim3(C / 32, C / 32), blk256, 0, stream>>>(Wg + (size_t)l * WCC, base + 3 * WCC, C, C);
    transpose_cvt_bf16<<<dim3(C / 32, C / 32), blk256, 0, stream>>>(Wo + (size_t)l * WCC, base + 4 * WCC, C, C);
    transpose_cvt_bf16<<<dim3(C / 32, FFN / 32), blk256, 0, stream>>>(W1 + (size_t)l * WCF, base + 5 * WCC, C, FFN);
    transpose_cvt_bf16<<<dim3(FFN / 32, C / 32), blk256, 0, stream>>>(W2 + (size_t)l * WCF, base + 5 * WCC + WCF, FFN, C);
  }

  copy_f32<<<dim3((BT * C + 255) / 256), blk256, 0, stream>>>(x_in, x, BT * C);

  const dim3 gCC(M / 128, C / 64);
  const dim3 gCF(M / 128, FFN / 64);

  for (int l = 0; l < L; ++l) {
    bf16* base = wpool + (size_t)l * LSTRIDE;
    const bf16 *Wq_t = base, *Wk_t = base + WCC, *Wv_t = base + 2 * WCC,
               *Wg_t = base + 3 * WCC, *Wo_t = base + 4 * WCC,
               *W1_t = base + 5 * WCC, *W2_t = base + 5 * WCC + WCF;
    const float* bq_l = bq + (size_t)l * C;   const float* bk_l = bk + (size_t)l * C;
    const float* bv_l = bv + (size_t)l * C;   const float* bg_l = bg + (size_t)l * C;
    const float* bo_l = bo + (size_t)l * C;
    const float* b1_l = b1 + (size_t)l * FFN; const float* b2_l = b2 + (size_t)l * C;

    // --- retention block ---
    ln_to_bf16<<<dim3(BT), blk256, 0, stream>>>(x, hb, C);
    gemm_bf16_wmma<<<gCC, blk256, 0, stream>>>(hb, Wq_t, bq_l, q32, nullptr, M, C, C);
    gemm_bf16_wmma<<<gCC, blk256, 0, stream>>>(hb, Wk_t, bk_l, k32, nullptr, M, C, C);
    gemm_bf16_wmma<<<gCC, blk256, 0, stream>>>(hb, Wv_t, bv_l, v32, nullptr, M, C, C);
    gemm_bf16_wmma<<<gCC, blk256, 0, stream>>>(hb, Wg_t, bg_l, g32, nullptr, M, C, C);
    rope_prep<<<dim3((B * T * H * (KD / 2)) / 256), blk256, 0, stream>>>(q32, k32, v32,
                                                                         qrb, krb, vrb);
    retention_wmma<<<dim3(T / 16, B * H), dim3(32), 0, stream>>>(qrb, krb, vrb, ret32);
    gate_ln<<<dim3(BT), blk256, 0, stream>>>(ret32, g32, gtb);
    gemm_bf16_wmma<<<gCC, blk256, 0, stream>>>(gtb, Wo_t, bo_l, x, x, M, C, C);  // + residual

    // --- FFN block ---
    ln_to_bf16<<<dim3(BT), blk256, 0, stream>>>(x, hb, C);
    gemm_bf16_wmma<<<gCF, blk256, 0, stream>>>(hb, W1_t, b1_l, f32buf, nullptr, M, FFN, C);
    ln_to_bf16<<<dim3(BT), blk256, 0, stream>>>(f32buf, hb, FFN);
    gemm_bf16_wmma<<<gCC, blk256, 0, stream>>>(hb, W2_t, b2_l, x, x, M, C, FFN);  // + residual
  }
  (void)in_sizes; (void)n_in; (void)out_size; (void)ws_size;
}
